// attentionLayer_72078141161995
// MI455X (gfx1250) — compile-verified, matrix operand order checked by reading, not run on hardware
//
#include <hip/hip_runtime.h>
#include <hip/hip_bf16.h>

// Problem sizes (fixed by the reference)
#define BB 4
#define CC 64
#define NN 4096      // H*W = 64*64
#define DKk 64
#define DVv 64

typedef __bf16 bf16;
typedef __bf16 v8bf  __attribute__((ext_vector_type(8)));
typedef __bf16 v16bf __attribute__((ext_vector_type(16)));
typedef float  v8f   __attribute__((ext_vector_type(8)));

// Build a 16x32 bf16 WMMA fragment for one lane from two contiguous 8-element
// (16-byte) runs.  Per the CDNA5 ISA (16-bit A 16x32 table): low lanes hold
// K = {0..7, 16..23}, high lanes K = {8..15, 24..31}; B mirrors with N per lane.
__device__ __forceinline__ v16bf make_frag(const bf16* lo, const bf16* hi) {
  v8bf a = *reinterpret_cast<const v8bf*>(lo);
  v8bf b = *reinterpret_cast<const v8bf*>(hi);
  return __builtin_shufflevector(a, b, 0, 1, 2, 3, 4, 5, 6, 7,
                                       8, 9, 10, 11, 12, 13, 14, 15);
}

// Post-softmax multiplicative mask value for (row n, col m), per _ordered_mask.
__device__ __forceinline__ float mask_val(int ordering, int n, int m) {
  const int L = NN - 1;
  switch (ordering) {
    case 1: return (m <= n)     ? 1.f : 0.f;  // tril
    case 2: return (L - m <= n) ? 1.f : 0.f;  // flip(tril, 1)
    case 3: return (L - n <= m) ? 1.f : 0.f;  // rot90(tril, 1)
    case 4: return (n <= m)     ? 1.f : 0.f;  // flip(rot90(tril,1), 0)
    case 5: return (n <= m)     ? 1.f : 0.f;  // rot90(tril, 2)
    case 6: return (n <= L - m) ? 1.f : 0.f;  // flip(rot90(tril,2), 1)
    case 7: return (n <= L - m) ? 1.f : 0.f;  // rot90(tril, 3)
    case 8: return (n <= m)     ? 1.f : 0.f;  // flip(rot90(tril,3), 1)
    default: return 1.f;                      // no mask
  }
}

// ---------------------------------------------------------------------------
// Kernel 1: QKV projection.  Q,K stored [B][N][64] bf16 (row per pixel),
// V stored transposed [B][64][N] bf16 (row per channel) so all attention
// WMMA fragments are contiguous 16B loads.
// ---------------------------------------------------------------------------
__global__ __launch_bounds__(256) void qkv_kernel(
    const float* __restrict__ x,
    const float* __restrict__ Wq, const float* __restrict__ bq,
    const float* __restrict__ Wk, const float* __restrict__ bk,
    const float* __restrict__ Wv, const float* __restrict__ bv,
    bf16* __restrict__ Qb, bf16* __restrict__ Kb, bf16* __restrict__ Vt) {
  __shared__ float xs[64 * 64];  // [c][p]
  const int nb64  = NN / 64;
  const int b     = blockIdx.x / nb64;
  const int pbase = (blockIdx.x % nb64) * 64;

  for (int idx = threadIdx.x; idx < 64 * 64; idx += 256) {
    const int c = idx >> 6, p = idx & 63;
    xs[c * 64 + p] = x[((size_t)b * CC + c) * NN + pbase + p];
  }
  __syncthreads();

  const int p = threadIdx.x & 63;
  const int g = threadIdx.x >> 6;  // 0..3 (uniform within each wave)
  const int n = pbase + p;

  for (int j = 0; j < 48; ++j) {
    const int r = g * 48 + j;  // 0..191 over stacked [Wq; Wk; Wv]
    const float* wrow;
    float bias;
    int kind, rr;
    if (r < 64)       { wrow = Wq + r * 64;         bias = bq[r];       kind = 0; rr = r; }
    else if (r < 128) { wrow = Wk + (r - 64) * 64;  bias = bk[r - 64];  kind = 1; rr = r - 64; }
    else              { wrow = Wv + (r - 128) * 64; bias = bv[r - 128]; kind = 2; rr = r - 128; }
    float acc = bias;
#pragma unroll 8
    for (int c = 0; c < 64; ++c) acc += wrow[c] * xs[c * 64 + p];
    const bf16 h = (bf16)acc;
    if (kind == 0)      Qb[((size_t)b * NN + n) * DKk + rr] = h;
    else if (kind == 1) Kb[((size_t)b * NN + n) * DKk + rr] = h;
    else                Vt[((size_t)b * DVv + rr) * NN + n] = h;
  }
}

// ---------------------------------------------------------------------------
// Kernel 2: streaming (flash-style) attention with post-softmax mask.
// 4 waves / block; each wave owns 16 query rows and sweeps all 4096 keys in
// chunks of 32, doing 4 WMMA for S = Q K^T and 4 WMMA for the P.V^T update.
// Denominator is UNMASKED (full softmax); mask multiplies the numerator only.
// ---------------------------------------------------------------------------
__global__ __launch_bounds__(128) void attn_kernel(
    const bf16* __restrict__ Qb, const bf16* __restrict__ Kb,
    const bf16* __restrict__ Vt, float* __restrict__ Ob,
    const int* __restrict__ ordp) {
  __shared__ bf16 psh[4 * 16 * 32];  // per-wave P staging (C-layout -> A-layout)

  const int ordering = *ordp;
  const int nb64 = NN / 64;
  const int b    = blockIdx.x / nb64;
  const int nblk = blockIdx.x % nb64;
  const int wave = threadIdx.x >> 5;
  const int lane = threadIdx.x & 31;
  const int l16  = lane & 15;
  const int hi8  = (lane >> 4) << 3;  // 0 for lanes 0-15, 8 for lanes 16-31
  const int nbase = nblk * 64 + wave * 16;

  // A-fragments of Q (rows nbase..nbase+15, dk split 0..31 / 32..63)
  const bf16* qrow = Qb + ((size_t)b * NN + nbase + l16) * DKk;
  const v16bf aq0 = make_frag(qrow + hi8,      qrow + 16 + hi8);
  const v16bf aq1 = make_frag(qrow + 32 + hi8, qrow + 48 + hi8);

  v8f acc0 = {}, acc1 = {}, acc2 = {}, acc3 = {};
  float mrow[8], lrow[8];
#pragma unroll
  for (int v = 0; v < 8; ++v) { mrow[v] = -3.0e38f; lrow[v] = 0.f; }

  bf16* mysh = psh + wave * (16 * 32);

  for (int mb = 0; mb < NN; mb += 32) {
    // ---- S tiles: two 16x16 f32 accumulators over dk = 64 (2 x K=32) ----
    v8f S0 = {}, S1 = {};
    {
      const bf16* kr0 = Kb + ((size_t)b * NN + mb + l16) * DKk;
      v16bf bk = make_frag(kr0 + hi8, kr0 + 16 + hi8);
      S0 = __builtin_amdgcn_wmma_f32_16x16x32_bf16(false, aq0, false, bk, (short)0, S0, false, false);
      bk = make_frag(kr0 + 32 + hi8, kr0 + 48 + hi8);
      S0 = __builtin_amdgcn_wmma_f32_16x16x32_bf16(false, aq1, false, bk, (short)0, S0, false, false);

      const bf16* kr1 = Kb + ((size_t)b * NN + mb + 16 + l16) * DKk;
      v16bf bk1 = make_frag(kr1 + hi8, kr1 + 16 + hi8);
      S1 = __builtin_amdgcn_wmma_f32_16x16x32_bf16(false, aq0, false, bk1, (short)0, S1, false, false);
      bk1 = make_frag(kr1 + 32 + hi8, kr1 + 48 + hi8);
      S1 = __builtin_amdgcn_wmma_f32_16x16x32_bf16(false, aq1, false, bk1, (short)0, S1, false, false);
    }

    // ---- online softmax: row = v + hi8 lives across the 16 lanes of a half ----
    float e0[8], e1[8], scale[8];
#pragma unroll
    for (int v = 0; v < 8; ++v) {
      float t = fmaxf(S0[v], S1[v]);
      t = fmaxf(t, __shfl_xor(t, 1, 32));
      t = fmaxf(t, __shfl_xor(t, 2, 32));
      t = fmaxf(t, __shfl_xor(t, 4, 32));
      t = fmaxf(t, __shfl_xor(t, 8, 32));
      const float mn = fmaxf(mrow[v], t);
      scale[v] = __expf(mrow[v] - mn);
      mrow[v] = mn;
      e0[v] = __expf(S0[v] - mn);
      e1[v] = __expf(S1[v] - mn);
      float s = e0[v] + e1[v];       // UNMASKED denominator
      s += __shfl_xor(s, 1, 32);
      s += __shfl_xor(s, 2, 32);
      s += __shfl_xor(s, 4, 32);
      s += __shfl_xor(s, 8, 32);
      lrow[v] = lrow[v] * scale[v] + s;
    }
#pragma unroll
    for (int v = 0; v < 8; ++v) {
      acc0[v] *= scale[v]; acc1[v] *= scale[v];
      acc2[v] *= scale[v]; acc3[v] *= scale[v];
    }

    __syncthreads();  // WAR: previous iteration's P fragment loads are done
    // ---- masked P -> LDS, row-major [16][32] (C-layout scatter) ----
#pragma unroll
    for (int v = 0; v < 8; ++v) {
      const int row = v + hi8;
      const int n   = nbase + row;
      mysh[row * 32 + l16]      = (bf16)(e0[v] * mask_val(ordering, n, mb + l16));
      mysh[row * 32 + 16 + l16] = (bf16)(e1[v] * mask_val(ordering, n, mb + 16 + l16));
    }
    __syncthreads();  // RAW: stores visible before A-layout reload

    // ---- P as A-fragment (16 rows x 32 K), then P . V^T (4 dv-tiles) ----
    const bf16* prow = mysh + l16 * 32;
    const v16bf pa = make_frag(prow + hi8, prow + 16 + hi8);

    const bf16* vr0 = Vt + ((size_t)b * DVv + 0  + l16) * NN + mb;
    const bf16* vr1 = Vt + ((size_t)b * DVv + 16 + l16) * NN + mb;
    const bf16* vr2 = Vt + ((size_t)b * DVv + 32 + l16) * NN + mb;
    const bf16* vr3 = Vt + ((size_t)b * DVv + 48 + l16) * NN + mb;
    v16bf bv;
    bv = make_frag(vr0 + hi8, vr0 + 16 + hi8);
    acc0 = __builtin_amdgcn_wmma_f32_16x16x32_bf16(false, pa, false, bv, (short)0, acc0, false, false);
    bv = make_frag(vr1 + hi8, vr1 + 16 + hi8);
    acc1 = __builtin_amdgcn_wmma_f32_16x16x32_bf16(false, pa, false, bv, (short)0, acc1, false, false);
    bv = make_frag(vr2 + hi8, vr2 + 16 + hi8);
    acc2 = __builtin_amdgcn_wmma_f32_16x16x32_bf16(false, pa, false, bv, (short)0, acc2, false, false);
    bv = make_frag(vr3 + hi8, vr3 + 16 + hi8);
    acc3 = __builtin_amdgcn_wmma_f32_16x16x32_bf16(false, pa, false, bv, (short)0, acc3, false, false);
  }

  // ---- epilogue: o = acc / l, stored [B][N][64] f32 ----
#pragma unroll
  for (int v = 0; v < 8; ++v) {
    const int n = nbase + v + hi8;
    const float inv = 1.0f / lrow[v];
    float* orow = Ob + ((size_t)b * NN + n) * DVv;
    orow[0  + l16] = acc0[v] * inv;
    orow[16 + l16] = acc1[v] * inv;
    orow[32 + l16] = acc2[v] * inv;
    orow[48 + l16] = acc3[v] * inv;
  }
}

// ---------------------------------------------------------------------------
// Kernel 3: out = Wo @ concat([o, x], channel) + bo      ([64 x 128] GEMM)
// ---------------------------------------------------------------------------
__global__ __launch_bounds__(256) void oproj_kernel(
    const float* __restrict__ Ob, const float* __restrict__ x,
    const float* __restrict__ Wo, const float* __restrict__ bo,
    float* __restrict__ out) {
  __shared__ float ocs[128 * 65];  // padded stride 65 to dodge bank conflicts
  const int nb64  = NN / 64;
  const int b     = blockIdx.x / nb64;
  const int pbase = (blockIdx.x % nb64) * 64;

  for (int idx = threadIdx.x; idx < 64 * 64; idx += 256) {
    const int p = idx >> 6, d = idx & 63;
    ocs[d * 65 + p] = Ob[((size_t)b * NN + pbase + p) * DVv + d];
  }
  for (int idx = threadIdx.x; idx < 64 * 64; idx += 256) {
    const int c = idx >> 6, p = idx & 63;
    ocs[(64 + c) * 65 + p] = x[((size_t)b * CC + c) * NN + pbase + p];
  }
  __syncthreads();

  const int p = threadIdx.x & 63;
  const int g = threadIdx.x >> 6;
  for (int j = 0; j < 16; ++j) {
    const int o = g * 16 + j;
    float acc = bo[o];
    const float* wrow = Wo + o * 128;
#pragma unroll 8
    for (int i = 0; i < 128; ++i) acc += wrow[i] * ocs[i * 65 + p];
    out[((size_t)b * DVv + o) * NN + pbase + p] = acc;
  }
}

// ---------------------------------------------------------------------------
extern "C" void kernel_launch(void* const* d_in, const int* in_sizes, int n_in,
                              void* d_out, int out_size, void* d_ws, size_t ws_size,
                              hipStream_t stream) {
  const float* x  = (const float*)d_in[0];
  const float* Wq = (const float*)d_in[1];
  const float* bq = (const float*)d_in[2];
  const float* Wk = (const float*)d_in[3];
  const float* bk = (const float*)d_in[4];
  const float* Wv = (const float*)d_in[5];
  const float* bv = (const float*)d_in[6];
  const float* Wo = (const float*)d_in[7];
  const float* bo = (const float*)d_in[8];
  const int*   od = (const int*)d_in[9];
  float* out = (float*)d_out;

  // Workspace layout: Qb(2MB) | Kb(2MB) | Vt(2MB) | Ob(4MB)  ~10MB total
  bf16* Qb = (bf16*)d_ws;
  bf16* Kb = Qb + (size_t)BB * NN * DKk;
  bf16* Vt = Kb + (size_t)BB * NN * DKk;
  float* Ob = (float*)(Vt + (size_t)BB * NN * DVv);

  const dim3 grid(BB * (NN / 64));  // 256 blocks
  qkv_kernel<<<grid, 256, 0, stream>>>(x, Wq, bq, Wk, bk, Wv, bv, Qb, Kb, Vt);
  attn_kernel<<<grid, 128, 0, stream>>>(Qb, Kb, Vt, Ob, od);
  oproj_kernel<<<grid, 256, 0, stream>>>(Ob, x, Wo, bo, out);
}
